// SocialLSTM_74242804678679
// MI455X (gfx1250) — compile-verified, matrix operand order checked by reading, not run on hardware
//
#include <hip/hip_runtime.h>
#include <hip/hip_bf16.h>

// ---------------------------------------------------------------------------
// Social LSTM on MI455X (gfx1250, wave32, WMMA).
// Per step, ONE fused kernel computes gates = e@Wih^T + h@Whh^T + b via
// v_wmma_f32_16x16x32_f16, exchanges the 4 gate tiles through LDS, and applies
// the LSTM cell pointwise (writes c, h_f32, h_f16).  No gates buffer, no
// separate cell kernel: saves ~60 MB of L2 traffic and one launch per step on
// the sequential critical path.
// ---------------------------------------------------------------------------

#define BN   4096
#define TT   32
#define POSD 2
#define SOCD 16
#define EMBD 512
#define HIDD 512
#define GATE 2048   // 4*HID
#define PRED 12

// LDS gate-exchange tile: 4 gates x 128 rows x (32 j + 4 pad) floats.
// Stride 36 => the lane16-31 half of a C-fragment store (row offset +8) lands
// on a disjoint bank group from lanes 0-15 (8*36 % 64 == 32).
#define LDS_STRIDE 36
#define LDS_GATE   (128 * LDS_STRIDE)
#define LDS_BYTES  (4 * LDS_GATE * 4)

typedef __attribute__((ext_vector_type(8)))  _Float16 v8h;
typedef __attribute__((ext_vector_type(16))) _Float16 v16h;
typedef __attribute__((ext_vector_type(8)))  float    v8f;

// ---------------------------------------------------------------------------
// Fragment loader: CDNA5 16-bit 16x32 A-matrix layout (ISA 7.12.2).
// lane = 16*khalf + r.  elements 0..7  = K[k0 + 8*khalf .. +7]
//                       elements 8..15 = K[k0 + 16 + 8*khalf .. +7]
// B (=W^T) uses the identical per-lane layout with r = column index; column n
// of W^T is row n of W, so all loads are contiguous global_load_b128s.
// ---------------------------------------------------------------------------
__device__ __forceinline__ v16h load_frag16(const _Float16* __restrict__ base,
                                            int row, int k0, int khalf) {
  const _Float16* p = base + (size_t)row * HIDD + k0 + khalf * 8;
  v8h lo = *(const v8h*)(p);
  v8h hi = *(const v8h*)(p + 16);
  return __builtin_shufflevector(lo, hi, 0,1,2,3,4,5,6,7,8,9,10,11,12,13,14,15);
}

__device__ __forceinline__ float sigf(float x) { return 1.f / (1.f + __expf(-x)); }

// ---------------------------------------------------------------------------
// Fused LSTM step.
//   Workgroup tile: 128 batch rows x 32 hidden columns x all 4 gates.
//   8 waves as (wr in 0..1, g in 0..3): wave computes gate g for a 64x32 tile
//   (4x2 wmma 16x16 tiles), weight rows = g*512 + j.
//   Two fused K-phases: A1=emb_f16 (B1=Wih_f16), A2=h_f16 (B2=Whh_f16).
//   Epilogue: acc -> LDS, barrier, pointwise LSTM cell, write c / h32 / h16.
// Grid must tile exactly (EXEC all-ones required for WMMA).
// ---------------------------------------------------------------------------
__global__ __launch_bounds__(256)
void lstm_step_fused_kernel(const _Float16* __restrict__ A1,
                            const _Float16* __restrict__ B1,
                            const _Float16* __restrict__ A2,
                            const _Float16* __restrict__ B2,
                            const float* __restrict__ bsum,   // bih+bhh [2048]
                            float* __restrict__ c32,
                            float* __restrict__ h32,
                            _Float16* __restrict__ h16) {
  extern __shared__ float sh[];          // LDS_BYTES
  const int lane  = threadIdx.x & 31;
  const int wave  = threadIdx.x >> 5;
  const int wr    = wave & 1;            // M half (0..1)
  const int g     = wave >> 1;           // gate index 0..3 (i,f,g,o)
  const int khalf = lane >> 4;
  const int lrow  = lane & 15;
  const int blockM = blockIdx.y * 128;
  const int jbase  = blockIdx.x * 32;    // hidden-column tile
  const int mbase  = blockM + wr * 64;
  const int wrow0  = g * HIDD + jbase;   // weight-matrix row base for gate g

  v8f acc[4][2];
#pragma unroll
  for (int mt = 0; mt < 4; ++mt)
#pragma unroll
    for (int nt = 0; nt < 2; ++nt)
      acc[mt][nt] = (v8f){0.f,0.f,0.f,0.f,0.f,0.f,0.f,0.f};

#pragma unroll
  for (int phase = 0; phase < 2; ++phase) {
    const _Float16* __restrict__ A = phase ? A2 : A1;
    const _Float16* __restrict__ W = phase ? B2 : B1;
#pragma unroll 2
    for (int k0 = 0; k0 < HIDD; k0 += 32) {
      v16h af[4], bf_[2];
#pragma unroll
      for (int mt = 0; mt < 4; ++mt)
        af[mt] = load_frag16(A, mbase + mt * 16 + lrow, k0, khalf);
#pragma unroll
      for (int nt = 0; nt < 2; ++nt)
        bf_[nt] = load_frag16(W, wrow0 + nt * 16 + lrow, k0, khalf);
#pragma unroll
      for (int mt = 0; mt < 4; ++mt)
#pragma unroll
        for (int nt = 0; nt < 2; ++nt)
          acc[mt][nt] = __builtin_amdgcn_wmma_f32_16x16x32_f16(
              /*neg_a=*/false, af[mt], /*neg_b=*/false, bf_[nt],
              /*c_mod=*/(short)0, acc[mt][nt],
              /*reuse_a=*/false, /*reuse_b=*/false);
    }
  }

  // ---- Gate exchange through LDS ----
  // C layout (ISA 7.12.2): VGPR r, lanes 0-15 -> M=r, lanes 16-31 -> M=8+r,
  // N = lane&15.
  float* shg = sh + g * LDS_GATE;
#pragma unroll
  for (int nt = 0; nt < 2; ++nt) {
    const int j = nt * 16 + lrow;
#pragma unroll
    for (int mt = 0; mt < 4; ++mt) {
      const int row0 = wr * 64 + mt * 16 + khalf * 8;
#pragma unroll
      for (int r = 0; r < 8; ++r)
        shg[(row0 + r) * LDS_STRIDE + j] = acc[mt][nt][r];
    }
  }
  __syncthreads();

  // ---- Pointwise LSTM cell: 128x32 elements, 16 per thread ----
  for (int e = threadIdx.x; e < 128 * 32; e += 256) {
    const int row = e >> 5;
    const int j   = e & 31;
    const int gj  = jbase + j;
    const int lo  = row * LDS_STRIDE + j;
    const float gi = sh[0 * LDS_GATE + lo] + bsum[gj];
    const float gf = sh[1 * LDS_GATE + lo] + bsum[HIDD + gj];
    const float gg = sh[2 * LDS_GATE + lo] + bsum[2 * HIDD + gj];
    const float go = sh[3 * LDS_GATE + lo] + bsum[3 * HIDD + gj];
    const size_t idx = (size_t)(blockM + row) * HIDD + gj;
    const float cn = sigf(gf) * c32[idx] + sigf(gi) * tanhf(gg);
    const float hn = sigf(go) * tanhf(cn);
    c32[idx] = cn;
    h32[idx] = hn;
    h16[idx] = (_Float16)hn;
  }
}

// ---------------------------------------------------------------------------
// Encoder embedding at time t: relu(x@Wp^T+bp) + relu(social@Ws^T+bs) -> f16
// ---------------------------------------------------------------------------
__global__ __launch_bounds__(256)
void encoder_emb_kernel(const float* __restrict__ x,
                        const float* __restrict__ social,
                        const float* __restrict__ Wp, const float* __restrict__ bp,
                        const float* __restrict__ Ws, const float* __restrict__ bs,
                        _Float16* __restrict__ emb16, int t) {
  const int idx = blockIdx.x * 256 + threadIdx.x;   // 0 .. BN*EMB-1
  const int b = idx >> 9;
  const int e = idx & 511;
  const float* xp = x + ((size_t)b * TT + t) * POSD;
  float pv = Wp[e * 2] * xp[0] + Wp[e * 2 + 1] * xp[1] + bp[e];
  const float* sp = social + ((size_t)b * TT + t) * SOCD;
  float sv = bs[e];
#pragma unroll
  for (int k = 0; k < SOCD; ++k) sv += Ws[e * SOCD + k] * sp[k];
  emb16[idx] = (_Float16)(fmaxf(pv, 0.f) + fmaxf(sv, 0.f));
}

// ---------------------------------------------------------------------------
// Decoder embedding: relu(dec_in@Wp^T+bp) + relu(bs)   (social input is zero)
// First step reads x[:, T-1, :], else the previous predicted step.
// ---------------------------------------------------------------------------
__global__ __launch_bounds__(256)
void decoder_emb_kernel(const float* __restrict__ dec_in,
                        const float* __restrict__ x, int first,
                        const float* __restrict__ Wp, const float* __restrict__ bp,
                        const float* __restrict__ bs,
                        _Float16* __restrict__ emb16) {
  const int idx = blockIdx.x * 256 + threadIdx.x;
  const int b = idx >> 9;
  const int e = idx & 511;
  float d0, d1;
  if (first) {
    const float* xp = x + ((size_t)b * TT + (TT - 1)) * POSD;
    d0 = xp[0]; d1 = xp[1];
  } else {
    d0 = dec_in[b * 2]; d1 = dec_in[b * 2 + 1];
  }
  const float pv = Wp[e * 2] * d0 + Wp[e * 2 + 1] * d1 + bp[e];
  emb16[idx] = (_Float16)(fmaxf(pv, 0.f) + fmaxf(bs[e], 0.f));
}

// ---------------------------------------------------------------------------
// Output head: step = h@Wf^T + bf.  One wave (32 lanes) per batch row,
// both output columns reduced with a __shfl_xor tree (wave32).
// ---------------------------------------------------------------------------
__global__ __launch_bounds__(256)
void out_proj_kernel(const float* __restrict__ h32,
                     const float* __restrict__ Wf, const float* __restrict__ bf,
                     float* __restrict__ out, float* __restrict__ dec_in, int s) {
  const int b    = (blockIdx.x * 256 + threadIdx.x) >> 5;   // wave id = row
  const int lane = threadIdx.x & 31;
  if (b >= BN) return;
  const float* hr = h32 + (size_t)b * HIDD;
  float a0 = 0.f, a1 = 0.f;
#pragma unroll 4
  for (int j = lane; j < HIDD; j += 32) {
    const float hv = hr[j];
    a0 += hv * Wf[j];
    a1 += hv * Wf[HIDD + j];
  }
#pragma unroll
  for (int off = 16; off > 0; off >>= 1) {
    a0 += __shfl_xor(a0, off, 32);
    a1 += __shfl_xor(a1, off, 32);
  }
  if (lane == 0) {
    const float o0 = a0 + bf[0];
    const float o1 = a1 + bf[1];
    float* op = out + (size_t)b * (PRED * POSD) + s * POSD;
    op[0] = o0; op[1] = o1;
    dec_in[b * 2]     = o0;
    dec_in[b * 2 + 1] = o1;
  }
}

// ---------------------------------------------------------------------------
// One-time: weights f32 -> f16 (4 MB total, stays L2-resident) + bias sum.
// ---------------------------------------------------------------------------
__global__ __launch_bounds__(256)
void convert_weights_kernel(const float* __restrict__ Wih,
                            const float* __restrict__ Whh,
                            const float* __restrict__ bih,
                            const float* __restrict__ bhh,
                            _Float16* __restrict__ Wih16,
                            _Float16* __restrict__ Whh16,
                            float* __restrict__ bsum) {
  const int idx = blockIdx.x * 256 + threadIdx.x;   // 0 .. GATE*HID-1
  Wih16[idx] = (_Float16)Wih[idx];
  Whh16[idx] = (_Float16)Whh[idx];
  if (idx < GATE) bsum[idx] = bih[idx] + bhh[idx];
}

__global__ __launch_bounds__(256)
void init_state_kernel(float* __restrict__ h32, float* __restrict__ c32,
                       _Float16* __restrict__ h16) {
  const int idx = blockIdx.x * 256 + threadIdx.x;   // 0 .. BN*HID-1
  h32[idx] = 0.f;
  c32[idx] = 0.f;
  h16[idx] = (_Float16)0.f;
}

// ---------------------------------------------------------------------------
// Launch sequence (graph-capture safe: stream-only, workspace scratch).
// ---------------------------------------------------------------------------
extern "C" void kernel_launch(void* const* d_in, const int* in_sizes, int n_in,
                              void* d_out, int out_size, void* d_ws, size_t ws_size,
                              hipStream_t stream) {
  (void)in_sizes; (void)n_in; (void)out_size; (void)ws_size;

  const float* x      = (const float*)d_in[0];
  const float* social = (const float*)d_in[1];
  const float* Wp     = (const float*)d_in[2];
  const float* bp     = (const float*)d_in[3];
  const float* Ws     = (const float*)d_in[4];
  const float* bs     = (const float*)d_in[5];
  const float* Wih    = (const float*)d_in[6];
  const float* Whh    = (const float*)d_in[7];
  const float* bih    = (const float*)d_in[8];
  const float* bhh    = (const float*)d_in[9];
  const float* Wf     = (const float*)d_in[10];
  const float* bf     = (const float*)d_in[11];
  float* out          = (float*)d_out;

  // Workspace carve-up (~28 MB, offsets naturally 256B aligned).
  char* ws = (char*)d_ws;
  size_t off = 0;
  _Float16* Wih16 = (_Float16*)(ws + off); off += (size_t)GATE * HIDD * sizeof(_Float16);
  _Float16* Whh16 = (_Float16*)(ws + off); off += (size_t)GATE * HIDD * sizeof(_Float16);
  _Float16* emb16 = (_Float16*)(ws + off); off += (size_t)BN * EMBD * sizeof(_Float16);
  _Float16* h16   = (_Float16*)(ws + off); off += (size_t)BN * HIDD * sizeof(_Float16);
  float*    h32   = (float*)(ws + off);    off += (size_t)BN * HIDD * sizeof(float);
  float*    c32   = (float*)(ws + off);    off += (size_t)BN * HIDD * sizeof(float);
  float*    bsum  = (float*)(ws + off);    off += (size_t)GATE * sizeof(float);
  float*    decin = (float*)(ws + off);    off += (size_t)BN * POSD * sizeof(float);

  const dim3 step_grid(HIDD / 32, BN / 128);    // 16 x 32 workgroups
  const dim3 step_blk(256);
  const int elem_blocks  = (BN * HIDD) / 256;   // 8192
  const int wconv_blocks = (GATE * HIDD) / 256; // 4096
  const int oproj_blocks = (BN * 32) / 256;     // one wave per row -> 512

  convert_weights_kernel<<<wconv_blocks, 256, 0, stream>>>(Wih, Whh, bih, bhh,
                                                           Wih16, Whh16, bsum);
  init_state_kernel<<<elem_blocks, 256, 0, stream>>>(h32, c32, h16);

  // ---- Encoder: 32 sequential LSTM steps ----
  for (int t = 0; t < TT; ++t) {
    encoder_emb_kernel<<<elem_blocks, 256, 0, stream>>>(x, social, Wp, bp, Ws, bs,
                                                        emb16, t);
    lstm_step_fused_kernel<<<step_grid, step_blk, LDS_BYTES, stream>>>(
        emb16, Wih16, h16, Whh16, bsum, c32, h32, h16);
  }

  // ---- Decoder: 12 sequential predicted steps ----
  for (int s = 0; s < PRED; ++s) {
    decoder_emb_kernel<<<elem_blocks, 256, 0, stream>>>(decin, x, (s == 0) ? 1 : 0,
                                                        Wp, bp, bs, emb16);
    lstm_step_fused_kernel<<<step_grid, step_blk, LDS_BYTES, stream>>>(
        emb16, Wih16, h16, Whh16, bsum, c32, h32, h16);
    out_proj_kernel<<<oproj_blocks, 256, 0, stream>>>(h32, Wf, bf, out, decin, s);
  }
}
